// QBBLinear_2276332667469
// MI455X (gfx1250) — matrix operand order repaired and checked by reading, not compiled, not run on hardware
//
#include <hip/hip_runtime.h>
#include <cstdint>
#include <cstddef>

// ---------------------------------------------------------------------------
// QBBLinear on gfx1250:  y = x @ (sum_k alpha_k * bases_k)^T
//   M = 4*2048 = 8192, N = OUT = 4096, Kdim = IN = 4096
// Pass 0 (if ws fits): convert x fp32 -> bf16 once            (64 MB in d_ws)
// Pass 1: fold alphas into +-1 bases -> W_q in bf16           (32 MB in d_ws)
// Pass 2: tiled bf16 WMMA GEMM (block 128x256, wave tile 64x64, BK=32),
//         fp32 accumulate. A-side staging is a pure bf16 copy when Pass 0
//         ran, else converts fp32->bf16 on the fly.
// ---------------------------------------------------------------------------

#define KBASES 4
#define OUTDIM 4096
#define INDIM  4096
#define MROWS  8192

typedef __attribute__((ext_vector_type(16))) __bf16  v16bf;
typedef __attribute__((ext_vector_type(8)))  __bf16  v8bf;
typedef __attribute__((ext_vector_type(8)))  float   v8f;
typedef __attribute__((ext_vector_type(4)))  float   v4f;

// ---------------------------------------------------------------------------
// Pass 0: x fp32 -> bf16, 8 elements per thread.
// ---------------------------------------------------------------------------
__global__ void qbb_convert_x(const float* __restrict__ x,
                              __bf16*      __restrict__ xbf) {
    size_t idx = ((size_t)blockIdx.x * blockDim.x + threadIdx.x) * 8;
    if (idx >= (size_t)MROWS * INDIM) return;
    v4f a = *(const v4f*)(x + idx);
    v4f b = *(const v4f*)(x + idx + 4);
    v8bf r;
#pragma unroll
    for (int j = 0; j < 4; ++j) {
        r[j]     = (__bf16)a[j];
        r[4 + j] = (__bf16)b[j];
    }
    *(v8bf*)(xbf + idx) = r;
}

// ---------------------------------------------------------------------------
// Pass 1: W_q[o,i] = sum_k alpha[k,o] * bases[k,o,i]  ->  bf16, row-major
// 8 elements per thread.
// ---------------------------------------------------------------------------
__global__ void qbb_build_wq(const int8_t* __restrict__ bases,
                             const float*  __restrict__ alphas,
                             __bf16*       __restrict__ wq) {
    const int VEC = 8;
    size_t idx = (size_t)blockIdx.x * blockDim.x + threadIdx.x;
    if (idx >= (size_t)OUTDIM * (INDIM / VEC)) return;
    int o  = (int)(idx / (INDIM / VEC));
    int i0 = (int)(idx % (INDIM / VEC)) * VEC;

    float a[KBASES];
#pragma unroll
    for (int k = 0; k < KBASES; ++k) a[k] = alphas[k * OUTDIM + o];

    float acc[VEC];
#pragma unroll
    for (int j = 0; j < VEC; ++j) acc[j] = 0.0f;

#pragma unroll
    for (int k = 0; k < KBASES; ++k) {
        const int8_t* p = bases + ((size_t)k * OUTDIM + o) * INDIM + i0;
        uint2 raw = *(const uint2*)p;            // 8 int8 in {-1,+1}
#pragma unroll
        for (int j = 0; j < 4; ++j) {
            float b0 = (float)(int8_t)(raw.x >> (8 * j));
            float b1 = (float)(int8_t)(raw.y >> (8 * j));
            acc[j]     = fmaf(a[k], b0, acc[j]);
            acc[4 + j] = fmaf(a[k], b1, acc[4 + j]);
        }
    }

    v8bf r;
#pragma unroll
    for (int j = 0; j < VEC; ++j) r[j] = (__bf16)acc[j];
    *(v8bf*)(wq + (size_t)o * INDIM + i0) = r;
}

// ---------------------------------------------------------------------------
// Pass 2: GEMM  out[m, o] = sum_i x[m,i] * wq[o,i]
// Block tile 128(M) x 256(N), BK=32, 256 threads = 8 waves (wave32).
// Wave grid 2(M) x 4(N); wave tile 64x64 -> 4x4 = 16 WMMA 16x16x32 per K-step.
// ---------------------------------------------------------------------------
#define BM  128
#define BN  256
#define BK  32
#define BKP 40   // padded LDS row length (bf16): 80 B stride -> conflict-free b128

union Frag { v16bf v; struct { v8bf lo, hi; } p; };

template <bool A_IS_BF16>
__global__ __launch_bounds__(256)
void qbb_gemm_bf16(const void*   __restrict__ xsrc,
                   const __bf16* __restrict__ wq,
                   float*        __restrict__ out) {
    __shared__ __bf16 As[BM * BKP];   // x tile   (10 KB)
    __shared__ __bf16 Bs[BN * BKP];   // wq tile  (20 KB)

    const int tid   = threadIdx.x;
    const int lane  = tid & 31;
    const int wave  = tid >> 5;
    const int waveM = wave >> 2;      // 0..1
    const int waveN = wave & 3;       // 0..3
    const int lr    = lane & 15;      // row / col within 16x16 tile
    const int lhi   = lane >> 4;      // half-wave selector

    const int blockN0 = blockIdx.x * BN;
    const int blockM0 = blockIdx.y * BM;

    // A staging: 2 threads per tile row, 16 elements each
    const int arow = tid >> 1;              // 0..127
    const int akof = (tid & 1) * 16;        // 0 or 16
    // B staging: 1 thread per tile row, full 32 bf16
    const int brow = tid;                   // 0..255

    v8f c[4][4];
#pragma unroll
    for (int mi = 0; mi < 4; ++mi)
#pragma unroll
        for (int ni = 0; ni < 4; ++ni)
#pragma unroll
            for (int v = 0; v < 8; ++v) c[mi][ni][v] = 0.0f;

    for (int kt = 0; kt < INDIM; kt += BK) {
        // clamped next-tile offset for prefetch (last iter re-prefetches self)
        const int knext = (kt + BK < INDIM) ? BK : 0;

        // ---- global -> registers (overlaps the barrier wait) ----
        v8bf alo, ahi;
        if constexpr (A_IS_BF16) {
            const __bf16* xp = (const __bf16*)xsrc
                             + (size_t)(blockM0 + arow) * INDIM + kt + akof;
            alo = *(const v8bf*)(xp + 0);
            ahi = *(const v8bf*)(xp + 8);
            __builtin_prefetch(xp + knext, 0, 0);      // global_prefetch_b8
        } else {
            const float* xp = (const float*)xsrc
                            + (size_t)(blockM0 + arow) * INDIM + kt + akof;
            v4f xa0 = *(const v4f*)(xp + 0);
            v4f xa1 = *(const v4f*)(xp + 4);
            v4f xa2 = *(const v4f*)(xp + 8);
            v4f xa3 = *(const v4f*)(xp + 12);
            __builtin_prefetch(xp + knext, 0, 0);
#pragma unroll
            for (int j = 0; j < 4; ++j) {              // v_cvt_pk_bf16_f32
                alo[j]     = (__bf16)xa0[j];
                alo[4 + j] = (__bf16)xa1[j];
                ahi[j]     = (__bf16)xa2[j];
                ahi[4 + j] = (__bf16)xa3[j];
            }
        }

        const __bf16* wp = wq + (size_t)(blockN0 + brow) * INDIM + kt;
        v8bf wb0 = *(const v8bf*)(wp + 0);
        v8bf wb1 = *(const v8bf*)(wp + 8);
        v8bf wb2 = *(const v8bf*)(wp + 16);
        v8bf wb3 = *(const v8bf*)(wp + 24);
        __builtin_prefetch(wp + knext, 0, 0);

        __syncthreads();   // previous iteration's LDS reads complete

        // ---- stage into LDS ----
        *(v8bf*)&As[arow * BKP + akof + 0] = alo;
        *(v8bf*)&As[arow * BKP + akof + 8] = ahi;
        *(v8bf*)&Bs[brow * BKP +  0] = wb0;
        *(v8bf*)&Bs[brow * BKP +  8] = wb1;
        *(v8bf*)&Bs[brow * BKP + 16] = wb2;
        *(v8bf*)&Bs[brow * BKP + 24] = wb3;

        __syncthreads();   // tile visible to all waves

        // ---- build fragments per ISA VGPR layouts ----
        // A 16x32 bf16: lane holds row lr, K chunks [lhi*8 .. +7], [+16 .. +23]
        Frag af[4];
        const int akb = lhi * 8;
#pragma unroll
        for (int mi = 0; mi < 4; ++mi) {
            const int r = waveM * 64 + mi * 16 + lr;
            af[mi].p.lo = *(const v8bf*)&As[r * BKP + akb + 0];
            af[mi].p.hi = *(const v8bf*)&As[r * BKP + akb + 16];
        }
        // B 32x16 bf16: lane holds column lr, K chunks [lhi*8 .. +7], [+16 .. +23]
        Frag bfr[4];
        const int bkb = lhi * 8;
#pragma unroll
        for (int ni = 0; ni < 4; ++ni) {
            const int n = waveN * 64 + ni * 16 + lr;
            bfr[ni].p.lo = *(const v8bf*)&Bs[n * BKP + bkb + 0];
            bfr[ni].p.hi = *(const v8bf*)&Bs[n * BKP + bkb + 16];
        }

        // ---- 16x v_wmma_f32_16x16x32_bf16 ----
#pragma unroll
        for (int mi = 0; mi < 4; ++mi)
#pragma unroll
            for (int ni = 0; ni < 4; ++ni)
                c[mi][ni] = __builtin_amdgcn_wmma_f32_16x16x32_bf16(
                    false, af[mi].v, false, bfr[ni].v,
                    (short)0, c[mi][ni], false, false);
    }

    // ---- store D: VGPR v -> M = v (lanes 0-15) / v+8 (lanes 16-31), N = lr ----
#pragma unroll
    for (int mi = 0; mi < 4; ++mi) {
#pragma unroll
        for (int ni = 0; ni < 4; ++ni) {
            const int col     = blockN0 + waveN * 64 + ni * 16 + lr;
            const int rowbase = blockM0 + waveM * 64 + mi * 16 + lhi * 8;
#pragma unroll
            for (int v = 0; v < 8; ++v)
                out[(size_t)(rowbase + v) * OUTDIM + col] = c[mi][ni][v];
        }
    }
}

// ---------------------------------------------------------------------------
// Host launcher
// ---------------------------------------------------------------------------
extern "C" void kernel_launch(void* const* d_in, const int* in_sizes, int n_in,
                              void* d_out, int out_size, void* d_ws, size_t ws_size,
                              hipStream_t stream) {
    const float*  x      = (const float*)d_in[0];
    const int8_t* bases  = (const int8_t*)d_in[1];   // int8 in {-1,+1}
    const float*  alphas = (const float*)d_in[2];
    float*        out    = (float*)d_out;

    const size_t wq_bytes  = (size_t)OUTDIM * INDIM * sizeof(__bf16);   // 32 MB
    const size_t xbf_bytes = (size_t)MROWS  * INDIM * sizeof(__bf16);   // 64 MB

    __bf16* wq  = (__bf16*)d_ws;
    __bf16* xbf = (__bf16*)((char*)d_ws + wq_bytes);
    const bool pre_convert_x = (ws_size >= wq_bytes + xbf_bytes);       // fixed per run

    // Pass 1: fold alphas into bases -> W_q bf16
    {
        const long long nvec = (long long)OUTDIM * (INDIM / 8);
        const int blk = 256;
        const int grd = (int)((nvec + blk - 1) / blk);
        qbb_build_wq<<<grd, blk, 0, stream>>>(bases, alphas, wq);
    }

    // Pass 0 (optional): x fp32 -> bf16 once, halves A-side L2 traffic and
    // removes conversions from the GEMM hot loop.
    if (pre_convert_x) {
        const long long nvec = (long long)MROWS * (INDIM / 8);
        const int blk = 256;
        const int grd = (int)((nvec + blk - 1) / blk);
        qbb_convert_x<<<grd, blk, 0, stream>>>(x, xbf);
    }

    // Pass 2: WMMA GEMM. Grid x = N blocks (fastest) so one M-stripe of x and
    // all of W_q stay L2-resident across the 16 N-blocks.
    dim3 grid(OUTDIM / BN, MROWS / BM);   // (16, 64)
    if (pre_convert_x) {
        qbb_gemm_bf16<true><<<grid, 256, 0, stream>>>((const void*)xbf, wq, out);
    } else {
        qbb_gemm_bf16<false><<<grid, 256, 0, stream>>>((const void*)x, wq, out);
    }
}
